// MultiHeadAttention_3753801417043
// MI455X (gfx1250) — matrix operand order repaired and checked by reading, not compiled, not run on hardware
//
#include <hip/hip_runtime.h>
#include <hip/hip_bf16.h>
#include <math.h>

// ---------------- CDNA5 WMMA plumbing ----------------
typedef __attribute__((ext_vector_type(16))) _Float16 v16h;
typedef __attribute__((ext_vector_type(8)))  _Float16 v8h;
typedef __attribute__((ext_vector_type(8)))  float    v8f;

union AFrag { v16h v; v8h h8[2]; };

// A-matrix fragment (16 x 32, f16), source row-major [M][ld].
// ISA 7.12.2: lane L<16 -> row M=L, halves 0..7 = K 0..7, halves 8..15 = K 16..23;
//             lane L>=16 -> same rows, K offset +8.  Two contiguous 16B loads/lane.
__device__ __forceinline__ v16h load_frag_a(const _Float16* src, int ld,
                                            int row0, int k0, int lane) {
    int m  = row0 + (lane & 15);
    int kb = k0 + ((lane >> 4) << 3);
    const _Float16* p = src + (size_t)m * ld + kb;
    AFrag f;
    f.h8[0] = *(const v8h*)(p);
    f.h8[1] = *(const v8h*)(p + 16);
    return f.v;
}

// B-matrix fragment (32 x 16, f16), source stored TRANSPOSED: row-major [N][ld]
// so B[k][n] = src[n*ld + k].  Same K order as A: lane L<16 -> K 0..7, 16..23;
// lane L>=16 -> K offset +8.  Two contiguous 16B loads/lane.
__device__ __forceinline__ v16h load_frag_b(const _Float16* src, int ld,
                                            int n0, int k0, int lane) {
    int n  = n0 + (lane & 15);
    int kb = k0 + ((lane >> 4) << 3);
    const _Float16* p = src + (size_t)n * ld + kb;
    AFrag f;
    f.h8[0] = *(const v8h*)(p);
    f.h8[1] = *(const v8h*)(p + 16);
    return f.v;
}

__device__ __forceinline__ v8f wmma_f16(v16h a, v16h b, v8f c) {
    return __builtin_amdgcn_wmma_f32_16x16x32_f16(false, a, false, b,
                                                  (short)0, c, false, false);
}

#define NB   32      // batches
#define SS   512     // sequence length == dim_model
#define DIN  128     // input feature dim
#define MTOT (NB * SS)       // 16384 rows of q/k/v
#define JT   32      // query-tile rows per workgroup in attention kernel

// ---------------- prep kernels ----------------
__global__ void convert_x_kernel(const float* __restrict__ x,
                                 _Float16* __restrict__ Xh, int n) {
    int i = blockIdx.x * blockDim.x + threadIdx.x;
    if (i < n) Xh[i] = (_Float16)x[i];
}

// WT[mat][n][k] = sum_h W[(h*128 + k)*512 + n]   (folds concatenate([x]*8))
__global__ void reduce_w_kernel(const float* __restrict__ Wq,
                                const float* __restrict__ Wk,
                                const float* __restrict__ Wv,
                                _Float16* __restrict__ WT) {
    int idx = blockIdx.x * blockDim.x + threadIdx.x;
    if (idx >= 3 * SS * DIN) return;
    int mat = idx >> 16;            // 512*128 = 65536 per matrix
    int rem = idx & 65535;
    int n   = rem >> 7;
    int kc  = rem & 127;
    const float* W = (mat == 0) ? Wq : (mat == 1) ? Wk : Wv;
    float s = 0.f;
#pragma unroll
    for (int h = 0; h < 8; ++h) s += W[(size_t)(h * DIN + kc) * SS + n];
    WT[idx] = (_Float16)s;
}

// ---------------- QKV GEMM: [16384,128] x [128,512] (x3) ----------------
__global__ __launch_bounds__(256) void qkv_kernel(const _Float16* __restrict__ Xh,
                                                  const _Float16* __restrict__ WT,
                                                  const float* __restrict__ bq,
                                                  const float* __restrict__ bk,
                                                  const float* __restrict__ bv,
                                                  _Float16* __restrict__ QKV) {
    const int mat = blockIdx.z;
    const _Float16* Wt   = WT + (size_t)mat * SS * DIN;      // [512][128] = B^T form
    const float*    bias = (mat == 0) ? bq : (mat == 1) ? bk : bv;
    _Float16*       Out  = QKV + (size_t)mat * MTOT * SS;

    const int tid = threadIdx.x, wave = tid >> 5, lane = tid & 31;
    const int m0 = blockIdx.x * 16;
    const int n0 = (blockIdx.y * 8 + wave) * 16;

    v8f acc = {0.f, 0.f, 0.f, 0.f, 0.f, 0.f, 0.f, 0.f};
#pragma unroll
    for (int ks = 0; ks < DIN / 32; ++ks) {
        v16h a = load_frag_a(Xh, DIN, m0, ks * 32, lane);
        v16h b = load_frag_b(Wt, DIN, n0, ks * 32, lane);
        acc = wmma_f16(a, b, acc);
    }
    const int n = n0 + (lane & 15);
    const float bn = bias[n];
#pragma unroll
    for (int r = 0; r < 8; ++r) {
        int m = m0 + r + ((lane >> 4) << 3);
        Out[(size_t)m * SS + n] = (_Float16)(acc[r] + bn);
    }
}

// ---------------- fused attention: scores -> softmax -> out = v @ attn^T ----
__global__ __launch_bounds__(256) void attn_kernel(const _Float16* __restrict__ Q,
                                                   const _Float16* __restrict__ K,
                                                   const _Float16* __restrict__ V,
                                                   float* __restrict__ out) {
    __shared__ float    sS[JT * SS];      // 64 KB scores (f32)
    __shared__ _Float16 sP[JT * SS];      // 32 KB probabilities (f16)
    __shared__ float    sMax[JT * 8];
    __shared__ float    sSum[JT * 8];

    const int b   = blockIdx.y;
    const int j0  = blockIdx.x * JT;
    const int tid = threadIdx.x;
    const int wave = tid >> 5, lane = tid & 31;
    const int jw = (wave & 1) * 16;       // query sub-tile (rows of attn / cols of out)
    const int nh = (wave >> 1) * 128;     // key / output-row quarter

    const _Float16* Qb = Q + (size_t)b * SS * SS;
    const _Float16* Kb = K + (size_t)b * SS * SS;
    const _Float16* Vb = V + (size_t)b * SS * SS;

    // ---- GEMM1: scores[j, t] = (q[j,:] . k[t,:]) / 8 ----
    v8f acc[8];
#pragma unroll
    for (int t = 0; t < 8; ++t)
        acc[t] = (v8f){0.f, 0.f, 0.f, 0.f, 0.f, 0.f, 0.f, 0.f};

    for (int ks = 0; ks < SS / 32; ++ks) {
        v16h a = load_frag_a(Qb, SS, j0 + jw, ks * 32, lane);
#pragma unroll
        for (int t = 0; t < 8; ++t) {
            v16h bf = load_frag_b(Kb, SS, nh + t * 16, ks * 32, lane);
            acc[t] = wmma_f16(a, bf, acc[t]);
        }
    }
#pragma unroll
    for (int t = 0; t < 8; ++t)
#pragma unroll
        for (int r = 0; r < 8; ++r) {
            int j = jw + r + ((lane >> 4) << 3);
            int n = nh + t * 16 + (lane & 15);
            sS[j * SS + n] = acc[t][r] * 0.125f;     // 1/sqrt(64)
        }
    __syncthreads();

    // ---- softmax over each row of 512 (8 threads per row) ----
    {
        int r = tid >> 3, seg = tid & 7, c0 = seg * 64;
        float m = -1e30f;
        for (int c = 0; c < 64; ++c) m = fmaxf(m, sS[r * SS + c0 + c]);
        sMax[r * 8 + seg] = m;
        __syncthreads();
        float rm = sMax[r * 8];
#pragma unroll
        for (int e = 1; e < 8; ++e) rm = fmaxf(rm, sMax[r * 8 + e]);
        float sum = 0.f;
        for (int c = 0; c < 64; ++c) {
            float ex = __expf(sS[r * SS + c0 + c] - rm);
            sS[r * SS + c0 + c] = ex;
            sum += ex;
        }
        sSum[r * 8 + seg] = sum;
        __syncthreads();
        float tot = 0.f;
#pragma unroll
        for (int e = 0; e < 8; ++e) tot += sSum[r * 8 + e];
        float inv = 1.0f / tot;
        for (int c = 0; c < 64; ++c)
            sP[r * SS + c0 + c] = (_Float16)(sS[r * SS + c0 + c] * inv);
    }
    __syncthreads();

    // ---- GEMM2: outT[j, i] = sum_k attn[j,k] * V[i,k]  (V is [N][K] form) ----
    v8f acc2[8];
#pragma unroll
    for (int t = 0; t < 8; ++t)
        acc2[t] = (v8f){0.f, 0.f, 0.f, 0.f, 0.f, 0.f, 0.f, 0.f};

    const int ih = nh;                    // output-row quarter
    for (int ks = 0; ks < SS / 32; ++ks) {
        v16h a = load_frag_a(sP, SS, jw, ks * 32, lane);
#pragma unroll
        for (int t = 0; t < 8; ++t) {
            v16h bf = load_frag_b(Vb, SS, ih + t * 16, ks * 32, lane);
            acc2[t] = wmma_f16(a, bf, acc2[t]);
        }
    }
    float* outb = out + (size_t)b * SS * SS;
#pragma unroll
    for (int t = 0; t < 8; ++t)
#pragma unroll
        for (int r = 0; r < 8; ++r) {
            int j = j0 + jw + r + ((lane >> 4) << 3);
            int i = ih + t * 16 + (lane & 15);
            outb[(size_t)i * SS + j] = acc2[t][r];   // out[b,i,j] = outT[j,i]
        }
}

// ---------------- launch ----------------
extern "C" void kernel_launch(void* const* d_in, const int* in_sizes, int n_in,
                              void* d_out, int out_size, void* d_ws, size_t ws_size,
                              hipStream_t stream) {
    const float* x  = (const float*)d_in[0];
    const float* Wq = (const float*)d_in[1];
    const float* bq = (const float*)d_in[2];
    const float* Wk = (const float*)d_in[3];
    const float* bk = (const float*)d_in[4];
    const float* Wv = (const float*)d_in[5];
    const float* bv = (const float*)d_in[6];
    float* out = (float*)d_out;

    // workspace layout (f16): Xh[16384][128] | WT[3][512][128] | QKV[3][16384][512]
    _Float16* Xh  = (_Float16*)d_ws;
    _Float16* WT  = Xh + (size_t)MTOT * DIN;
    _Float16* QKV = WT + (size_t)3 * SS * DIN;

    {
        int n = MTOT * DIN;
        convert_x_kernel<<<(n + 255) / 256, 256, 0, stream>>>(x, Xh, n);
    }
    {
        int n = 3 * SS * DIN;
        reduce_w_kernel<<<(n + 255) / 256, 256, 0, stream>>>(Wq, Wk, Wv, WT);
    }
    qkv_kernel<<<dim3(MTOT / 16, SS / 128, 3), 256, 0, stream>>>(Xh, WT, bq, bk, bv, QKV);

    const _Float16* Qh = QKV;
    const _Float16* Kh = QKV + (size_t)MTOT * SS;
    const _Float16* Vh = QKV + (size_t)2 * MTOT * SS;
    attn_kernel<<<dim3(SS / JT, NB), 256, 0, stream>>>(Qh, Kh, Vh, out);
}
